// GNNAggregationWithAttentionModule_4947802325003
// MI455X (gfx1250) — matrix operand order repaired and checked by reading, not compiled, not run on hardware
//
#include <hip/hip_runtime.h>

typedef __attribute__((ext_vector_type(16))) _Float16 v16h;
typedef __attribute__((ext_vector_type(8)))  float    v8f;

#define DD 256          // embedding dim (fixed by reference)
#define NEG_SLOPE 0.2f
#define ET_STRIDE 40    // halves per LDS row (32 data + 8 pad -> 80B, 16B aligned, bank-friendly)
#define PF_DIST 256     // A-stream prefetch distance in floats (8 chunks = 1KB ahead)

// ---------------- prep: ETg[d][n] = (f16)E[n][d], zero-padded to etgStride ----------------
// block = 256 threads handles a 64(n) x 256(d) tile via LDS transpose.
__global__ __launch_bounds__(256) void prep_transpose_kernel(
    const float* __restrict__ E, _Float16* __restrict__ ETg, int Ntot, int etgStride)
{
    __shared__ __align__(16) _Float16 T2[256 * 72];
    const int t  = threadIdx.x;         // d index
    const int n0 = blockIdx.x * 64;
    #pragma unroll 8
    for (int k = 0; k < 64; ++k) {
        const int n = n0 + k;
        const float v = (n < Ntot) ? E[(size_t)n * DD + t] : 0.f;
        T2[t * 72 + k] = (_Float16)v;
    }
    __syncthreads();
    const _Float16* src = &T2[t * 72];
    _Float16* dst = ETg + (size_t)t * etgStride + n0;
    #pragma unroll
    for (int s = 0; s < 8; ++s)
        *(uint4*)(dst + s * 8) = *(const uint4*)(src + s * 8);
}

// ---------------- prep: n_score[n] = E[n]·w_n ----------------
__global__ __launch_bounds__(DD) void prep_n_kernel(
    const float* __restrict__ E, const float* __restrict__ fc_w,
    float* __restrict__ n_score)
{
    const int n = blockIdx.x;
    const int t = threadIdx.x;
    __shared__ float red[DD];
    red[t] = E[(size_t)n * DD + t] * fc_w[DD + t];
    __syncthreads();
    #pragma unroll
    for (int s = DD / 2; s >= 1; s >>= 1) {
        if (t < s) red[t] += red[t + s];
        __syncthreads();
    }
    if (t == 0) n_score[n] = red[0];
}

// ---------------- prep: c_score[b] = E[idx[b]]·w_c + fc_b ----------------
__global__ __launch_bounds__(DD) void prep_c_kernel(
    const float* __restrict__ E, const float* __restrict__ fc_w,
    const float* __restrict__ fc_b, const int* __restrict__ idx,
    float* __restrict__ c_score)
{
    const int b = blockIdx.x;
    const int t = threadIdx.x;
    const int g = idx[b];
    __shared__ float red[DD];
    red[t] = E[(size_t)g * DD + t] * fc_w[t];
    __syncthreads();
    #pragma unroll
    for (int s = DD / 2; s >= 1; s >>= 1) {
        if (t < s) red[t] += red[t + s];
        __syncthreads();
    }
    if (t == 0) c_score[b] = red[0] + fc_b[0];
}

__device__ __forceinline__ _Float16 att_gate(float aval, float s) {
    s = (s >= 0.f) ? s : NEG_SLOPE * s;                  // leaky relu on score
    return (aval > 0.f) ? (_Float16)s : (_Float16)0.f;   // adjacency mask
}

struct AU { union { float4 f4[4]; float f[16]; }; };

// Load this lane's 16 A values for chunk k0 (ISA 16-bit A-fragment K pattern).
__device__ __forceinline__ void load_au(AU& au, const float* __restrict__ Arow,
                                        int k0, int koff, int Ntot) {
    if (k0 + 32 <= Ntot) {
        const float* ap = Arow + k0 + koff;
        au.f4[0] = *(const float4*)(ap);
        au.f4[1] = *(const float4*)(ap + 4);
        au.f4[2] = *(const float4*)(ap + 16);
        au.f4[3] = *(const float4*)(ap + 20);
    } else {
        #pragma unroll
        for (int i = 0; i < 8; ++i) {
            const int ka = k0 + koff + i;
            const int kb = k0 + 16 + koff + i;
            au.f[i]     = (ka < Ntot) ? Arow[ka] : 0.f;
            au.f[8 + i] = (kb < Ntot) ? Arow[kb] : 0.f;
        }
    }
}

// ---------------- main fused kernel: out = (mask*att) @ E + center ----------------
// block = 128 threads = 4 waves; tile = 32 rows x 256 cols.
// wave w: mt = w>>1 selects 16-row subtile, nh = w&1 selects 128-col half.
__global__ __launch_bounds__(128) void gnn_main_kernel(
    const float*    __restrict__ A,
    const float*    __restrict__ E,
    const int*      __restrict__ idx,
    const _Float16* __restrict__ ETg,
    const float*    __restrict__ n_score,
    const float*    __restrict__ c_score,
    float*          __restrict__ out,
    int Ntot, int etgStride)
{
    __shared__ __align__(16) _Float16 ET[DD * ET_STRIDE]; // E chunk, transposed [n][k]
    __shared__ __align__(16) float    nsl[32];            // n_score chunk

    const int t    = threadIdx.x;
    const int w    = t >> 5;
    const int lane = t & 31;
    const int mt   = w >> 1;
    const int nh   = w & 1;
    const int m    = lane & 15;     // row within 16-row subtile / col within 16-col frag
    const int ksel = lane >> 4;     // half-wave K selector
    const int koff = ksel * 8;      // A-matrix K offset per ISA 16-bit A layout

    const int   brow = blockIdx.x * 32 + mt * 16 + m;
    const int   grow = idx[brow];
    const float* Arow = A + (size_t)grow * (size_t)Ntot;
    const float cs   = c_score[brow];

    v8f acc[8];
    const v8f vzero = {0.f, 0.f, 0.f, 0.f, 0.f, 0.f, 0.f, 0.f};
    #pragma unroll
    for (int j = 0; j < 8; ++j) acc[j] = vzero;

    const int nChunks = (Ntot + 31) >> 5;

    // software pipeline: A values for chunk 0 loaded up front
    AU au;
    load_au(au, Arow, 0, koff, Ntot);

    for (int c = 0; c < nChunks; ++c) {
        const int k0 = c << 5;
        __syncthreads();   // previous chunk's ET reads done

        // ---- stage E^T chunk into LDS: 2 rows/thread, 4x b128 each, no guards
        #pragma unroll
        for (int rep = 0; rep < 2; ++rep) {
            const int n = t + rep * 128;
            const uint4* sp = (const uint4*)(ETg + (size_t)n * etgStride + k0);
            const uint4 q0 = sp[0], q1 = sp[1], q2 = sp[2], q3 = sp[3];
            uint4* dp = (uint4*)&ET[n * ET_STRIDE];
            dp[0] = q0; dp[1] = q1; dp[2] = q2; dp[3] = q3;
        }

        // ---- n_score chunk (guard only in the single tail chunk)
        if (k0 + 32 <= Ntot) {
            if (t < 32) nsl[t] = n_score[k0 + t];
        } else {
            if (t < 32) nsl[t] = (k0 + t < Ntot) ? n_score[k0 + t] : 0.f;
        }

        // ---- prefetch A stream ahead; issue next chunk's A loads now (full chunk of slack)
        if (k0 + PF_DIST < Ntot)
            __builtin_prefetch(Arow + k0 + PF_DIST + koff, 0, 0);
        AU aun;
        if (c + 1 < nChunks) {
            load_au(aun, Arow, k0 + 32, koff, Ntot);
        } else {
            #pragma unroll
            for (int i = 0; i < 16; ++i) aun.f[i] = 0.f;
        }
        __syncthreads();

        // ---- build 16x32 f16 A-fragment (masked leaky-relu attention)
        union { float4 f4[4]; float f[16]; } nu;
        nu.f4[0] = *(const float4*)&nsl[koff];
        nu.f4[1] = *(const float4*)&nsl[koff + 4];
        nu.f4[2] = *(const float4*)&nsl[16 + koff];
        nu.f4[3] = *(const float4*)&nsl[20 + koff];
        union { v16h v; _Float16 h[16]; } af;
        #pragma unroll
        for (int i = 0; i < 16; ++i)
            af.h[i] = att_gate(au.f[i], cs + nu.f[i]);

        // ---- B fragments from LDS + 8 WMMAs (compiler pipelines loads vs WMMA)
        union { v16h v; uint4 q[2]; } bf[8];
        #pragma unroll
        for (int j = 0; j < 8; ++j) {
            const _Float16* bp = &ET[(nh * 128 + j * 16 + m) * ET_STRIDE + ksel * 8];
            bf[j].q[0] = *(const uint4*)bp;
            bf[j].q[1] = *(const uint4*)(bp + 16);
        }
        #pragma unroll
        for (int j = 0; j < 8; ++j)
            acc[j] = __builtin_amdgcn_wmma_f32_16x16x32_f16(
                false, af.v, false, bf[j].v, (short)0, acc[j], false, false);

        au = aun;
    }

    // ---- epilogue: add gathered center row, store f32 per ISA C/D layout
    int g2[8];
    #pragma unroll
    for (int r = 0; r < 8; ++r)
        g2[r] = idx[blockIdx.x * 32 + mt * 16 + r + ksel * 8];
    #pragma unroll
    for (int j = 0; j < 8; ++j) {
        const int coln = nh * 128 + j * 16 + m;
        #pragma unroll
        for (int r = 0; r < 8; ++r) {
            const int row = blockIdx.x * 32 + mt * 16 + r + ksel * 8;
            const float cen = E[(size_t)g2[r] * DD + coln];
            out[(size_t)row * DD + coln] = acc[j][r] + cen;
        }
    }
}

extern "C" void kernel_launch(void* const* d_in, const int* in_sizes, int n_in,
                              void* d_out, int out_size, void* d_ws, size_t ws_size,
                              hipStream_t stream) {
    (void)n_in; (void)out_size; (void)ws_size;
    const float* A    = (const float*)d_in[0];
    const float* E    = (const float*)d_in[1];
    const float* fc_w = (const float*)d_in[2];
    const float* fc_b = (const float*)d_in[3];
    const int*   idx  = (const int*)d_in[4];
    float*       out  = (float*)d_out;

    const int D = in_sizes[2] / 2;        // 256
    const int N = in_sizes[1] / D;        // 10000
    const int B = in_sizes[4];            // 8192

    const int etgStride = ((N + 63) / 64) * 64;   // 10048: 128B-aligned rows, zero-padded tail

    char* ws = (char*)d_ws;
    _Float16* ETg     = (_Float16*)ws;
    float*    n_score = (float*)(ws + (size_t)DD * etgStride * sizeof(_Float16));
    float*    c_score = n_score + N;

    prep_transpose_kernel<<<etgStride / 64, 256, 0, stream>>>(E, ETg, N, etgStride);
    prep_n_kernel<<<N, DD, 0, stream>>>(E, fc_w, n_score);
    prep_c_kernel<<<B, DD, 0, stream>>>(E, fc_w, fc_b, idx, c_score);
    gnn_main_kernel<<<B / 32, 128, 0, stream>>>(A, E, idx, ETg, n_score, c_score, out, N, etgStride);
}
